// MultiHeadAttention_46832323396134
// MI455X (gfx1250) — compile-verified, matrix operand order checked by reading, not run on hardware
//
#include <hip/hip_runtime.h>
#include <math.h>

// ---------------------------------------------------------------------------
// MultiHeadAttention forward for MI455X (gfx1250), wave32 + WMMA bf16.
// B=8, S=2048, D_MODEL=128, H=4, Dk=Dv=32.
// d_out = [ out (B*S*128 f32) | attn (B*H*S*S f32) ]
// Dominant cost: streaming 537 MB attn write -> NT B128 stores.
// ---------------------------------------------------------------------------

#define B_SZ 8
#define S_LEN 2048
#define DM 128
#define NH 4
#define DK 32
#define LPITCH 2056            // 2048 + 8 floats: LDS bank-conflict padding

typedef __attribute__((ext_vector_type(16))) __bf16 v16bf;
typedef __attribute__((ext_vector_type(8)))  __bf16 v8bf;
typedef __attribute__((ext_vector_type(8)))  float  v8f;
typedef __attribute__((ext_vector_type(4)))  float  v4f;

union V16U { v16bf v; v8bf h[2]; };

// D = A(16x32 bf16) * B(32x16 bf16) + C(16x16 f32) — one wave.
__device__ __forceinline__ v8f wmma_bf16(v16bf a, v16bf b, v8f c) {
  return __builtin_amdgcn_wmma_f32_16x16x32_bf16(false, a, false, b,
                                                 (short)0, c, false, false);
}

// A/B operand for one K=32 slab from a bf16 row: lane supplies K elements
// {koff..koff+7} in elems 0..7 and {16+koff..16+koff+7} in elems 8..15.
__device__ __forceinline__ v16bf load_ab(const __bf16* row, int koff) {
  V16U u;
  u.h[0] = *(const v8bf*)(row + koff);
  u.h[1] = *(const v8bf*)(row + 16 + koff);
  return u.v;
}

// Same, but converting from fp32 source on the fly.
__device__ __forceinline__ v16bf cvt_ab(const float* p0, const float* p1) {
  v16bf r;
#pragma unroll
  for (int i = 0; i < 8; ++i) {
    r[i]     = (__bf16)p0[i];
    r[8 + i] = (__bf16)p1[i];
  }
  return r;
}

// ---------------------------------------------------------------------------
// Kernel 0: transpose + convert wq,wk,wv,wo (each [128 in][128 out] fp32)
// into bf16 [out][in] so WMMA B operands are contiguous rows.
// ---------------------------------------------------------------------------
__global__ __launch_bounds__(256) void prep_weights(
    const float* __restrict__ wq, const float* __restrict__ wk,
    const float* __restrict__ wv, const float* __restrict__ wo,
    __bf16* __restrict__ wT) {
  int idx = blockIdx.x * 256 + threadIdx.x;     // 0 .. 4*16384-1
  int mat = idx >> 14;
  int e   = idx & 16383;
  int o   = e >> 7;
  int i   = e & 127;
  const float* w = (mat == 0) ? wq : (mat == 1) ? wk : (mat == 2) ? wv : wo;
  wT[(size_t)mat * 16384 + o * DM + i] = (__bf16)w[i * DM + o];
}

// ---------------------------------------------------------------------------
// Kernel 1: QKV projection. One WG (4 waves) per 16 rows of [B*S, 128].
// 24 output tiles (3 matrices x 8 n-tiles), 6 per wave, K-loop of 4 slabs.
// q,k -> bf16 [B,H,S,32]; v -> bf16 transposed [B,H,32,S].
// ---------------------------------------------------------------------------
__global__ __launch_bounds__(128) void qkv_proj(
    const float* __restrict__ Q, const float* __restrict__ K,
    const float* __restrict__ V, const __bf16* __restrict__ wT,
    const float* __restrict__ bq, const float* __restrict__ bk,
    const float* __restrict__ bv,
    __bf16* __restrict__ q_ws, __bf16* __restrict__ k_ws,
    __bf16* __restrict__ vT_ws) {
  const int r0   = blockIdx.x * 16;
  const int lane = threadIdx.x & 31;
  const int wave = threadIdx.x >> 5;
  const int mrow = lane & 15;
  const int koff = (lane >> 4) * 8;

#pragma unroll
  for (int j = 0; j < 6; ++j) {
    const int id  = wave * 6 + j;
    const int mat = id >> 3;                 // 0=q 1=k 2=v
    const int nt  = id & 7;                  // 16-col tile of 128
    const float* X = (mat == 0) ? Q : (mat == 1) ? K : V;
    const __bf16* WTm = wT + (size_t)mat * 16384;
    v8f acc = {};
#pragma unroll
    for (int kc = 0; kc < 4; ++kc) {
      const float* xrow = X + (size_t)(r0 + mrow) * DM + kc * 32;
      v16bf a = cvt_ab(xrow + koff, xrow + 16 + koff);
      const __bf16* wrow = WTm + (size_t)(nt * 16 + mrow) * DM + kc * 32;
      v16bf bmat = load_ab(wrow, koff);
      acc = wmma_bf16(a, bmat, acc);
    }
    const float* bias = (mat == 0) ? bq : (mat == 1) ? bk : bv;
    const int n    = nt * 16 + (lane & 15);
    const int mhi  = (lane >> 4) * 8;
#pragma unroll
    for (int r = 0; r < 8; ++r) {
      const int m   = r + mhi;
      const float v = acc[r] + bias[n];
      const int row = r0 + m;
      const int b   = row >> 11;             // /S
      const int s   = row & (S_LEN - 1);
      const int h   = n >> 5;
      const int d   = n & 31;
      const size_t bh = (size_t)(b * NH + h);
      if (mat == 0)       q_ws[(bh * S_LEN + s) * DK + d] = (__bf16)v;
      else if (mat == 1)  k_ws[(bh * S_LEN + s) * DK + d] = (__bf16)v;
      else                vT_ws[(bh * DK + d) * S_LEN + s] = (__bf16)v;
    }
  }
}

// ---------------------------------------------------------------------------
// Kernel 2: attention. One WG (4 waves) per (b,h,16-query tile).
// Scores (16 x 2048 f32) stay LDS-resident; softmax in LDS; normalized
// probabilities written to d_out once (NT B128 stores); P*V via bf16 WMMA.
// ---------------------------------------------------------------------------
__global__ __launch_bounds__(128) void attn_kernel(
    const __bf16* __restrict__ q_ws, const __bf16* __restrict__ k_ws,
    const __bf16* __restrict__ vT_ws, const unsigned char* __restrict__ mask,
    float* __restrict__ attn_out, float* __restrict__ ctx_ws) {
  __shared__ float sP[16 * LPITCH];          // 131584 B score/prob block
  __shared__ float sRed[4 * 2 * 8 * 32];     // 8 KB cross-wave PV reduce
  __shared__ float sStat[16 * 8];
  __shared__ float sRow[32];                 // [0..15]=rowmax, [16..31]=1/sum

  const int bid = blockIdx.x;
  const int qt  = bid & 127;
  const int bh  = bid >> 7;                  // b*NH + h
  const int b   = bh >> 2;
  const int h   = bh & 3;
  const int q0  = qt * 16;

  const int lane = threadIdx.x & 31;
  const int wave = threadIdx.x >> 5;
  const int mrow = lane & 15;
  const int koff = (lane >> 4) * 8;
  const int ncol = lane & 15;
  const int mhi  = (lane >> 4) * 8;

  // ---- Phase 1: scores = (Q K^T) * 1/sqrt(32), + mask -> LDS -------------
  const __bf16* qrow = q_ws + ((size_t)bh * S_LEN + q0 + mrow) * DK;
  const v16bf aq = load_ab(qrow, koff);
  for (int kt = wave; kt < S_LEN / 16; kt += 4) {
    const __bf16* krow = k_ws + ((size_t)bh * S_LEN + kt * 16 + mrow) * DK;
    v16bf bk16 = load_ab(krow, koff);
    v8f zero = {};
    v8f sc = wmma_bf16(aq, bk16, zero);
#pragma unroll
    for (int r = 0; r < 8; ++r) {
      const int m = r + mhi;
      float v = sc[r] * 0.17677669529663687f;   // 1/sqrt(Dk)
      const unsigned char mk = __builtin_nontemporal_load(
          &mask[((size_t)b * S_LEN + q0 + m) * S_LEN + kt * 16 + ncol]);
      if (mk) v = -1e9f;
      sP[m * LPITCH + kt * 16 + ncol] = v;
    }
  }
  __syncthreads();

  // ---- Phase 2: row softmax statistics (8 threads per row) ---------------
  const int row = threadIdx.x >> 3;
  const int jj  = threadIdx.x & 7;
  float mx = -3.4e38f;
  for (int c = jj; c < S_LEN; c += 8) mx = fmaxf(mx, sP[row * LPITCH + c]);
  sStat[row * 8 + jj] = mx;
  __syncthreads();
  if (jj == 0) {
    float m2 = sStat[row * 8];
#pragma unroll
    for (int t = 1; t < 8; ++t) m2 = fmaxf(m2, sStat[row * 8 + t]);
    sRow[row] = m2;
  }
  __syncthreads();
  const float rmax = sRow[row];
  float psum = 0.f;
  for (int c = jj; c < S_LEN; c += 8) {
    float p = __expf(sP[row * LPITCH + c] - rmax);
    sP[row * LPITCH + c] = p;
    psum += p;
  }
  sStat[row * 8 + jj] = psum;
  __syncthreads();
  if (jj == 0) {
    float s2 = 0.f;
#pragma unroll
    for (int t = 0; t < 8; ++t) s2 += sStat[row * 8 + t];
    sRow[16 + row] = 1.0f / s2;
  }
  __syncthreads();

  // ---- Phase 3: normalize in LDS + single NT B128 write of attn ----------
  // 16*2048 floats = 8192 float4s; each thread streams 64 of them.
  float* aout = attn_out + ((size_t)bh * S_LEN + q0) * S_LEN;
  for (int g = threadIdx.x; g < (16 * S_LEN) / 4; g += 128) {
    const int m  = g >> 9;                    // float4s per row = 512
    const int c  = (g & 511) * 4;
    const float inv = sRow[16 + m];
    float* lp = &sP[m * LPITCH + c];          // 16B aligned (8224*m + 16*c4)
    v4f p = *(const v4f*)lp;
    p *= inv;
    *(v4f*)lp = p;
    __builtin_nontemporal_store(p, (v4f*)(aout + (size_t)m * S_LEN + c));
  }
  __syncthreads();

  // ---- Phase 4: context = P (16x2048) @ V (2048x32) via bf16 WMMA --------
  v8f c0 = {}, c1 = {};
  for (int ch = wave; ch < S_LEN / 32; ch += 4) {
    const float* prow = &sP[mrow * LPITCH + ch * 32];
    v16bf a = cvt_ab(prow + koff, prow + 16 + koff);
    const __bf16* v0 = vT_ws + ((size_t)bh * DK + ncol) * S_LEN + ch * 32;
    c0 = wmma_bf16(a, load_ab(v0, koff), c0);
    const __bf16* v1 = vT_ws + ((size_t)bh * DK + 16 + ncol) * S_LEN + ch * 32;
    c1 = wmma_bf16(a, load_ab(v1, koff), c1);
  }
#pragma unroll
  for (int r = 0; r < 8; ++r) {
    sRed[((wave * 2 + 0) * 8 + r) * 32 + lane] = c0[r];
    sRed[((wave * 2 + 1) * 8 + r) * 32 + lane] = c1[r];
  }
  __syncthreads();
  // reduce 4 waves, scatter into [B,S,H*Dv] fp32 for the output projection
  for (int e = threadIdx.x; e < 512; e += 128) {
    const int m = e >> 5, d = e & 31;
    const int tile = d >> 4, n = d & 15;
    const int l = n + 16 * (m >> 3), r = m & 7;
    float acc = 0.f;
#pragma unroll
    for (int w = 0; w < 4; ++w) acc += sRed[((w * 2 + tile) * 8 + r) * 32 + l];
    ctx_ws[((size_t)b * S_LEN + q0 + m) * DM + h * DK + d] = acc;
  }
}

// ---------------------------------------------------------------------------
// Kernel 3: output projection + bias + residual + LayerNorm, fused.
// One WG (4 waves) per 16 rows; 8 n-tiles (2/wave), K-loop of 4 slabs.
// ---------------------------------------------------------------------------
__global__ __launch_bounds__(128) void out_ln(
    const float* __restrict__ ctx_ws, const __bf16* __restrict__ woT,
    const float* __restrict__ bo, const float* __restrict__ Qin,
    const float* __restrict__ gamma, const float* __restrict__ beta,
    float* __restrict__ out) {
  __shared__ float sO[16 * 132];
  __shared__ float sStat[16 * 8];
  __shared__ float sMu[16], sRs[16];

  const int r0   = blockIdx.x * 16;
  const int lane = threadIdx.x & 31;
  const int wave = threadIdx.x >> 5;
  const int mrow = lane & 15;
  const int koff = (lane >> 4) * 8;

#pragma unroll
  for (int t = 0; t < 2; ++t) {
    const int nt = wave * 2 + t;
    v8f acc = {};
#pragma unroll
    for (int kc = 0; kc < 4; ++kc) {
      const float* xrow = ctx_ws + (size_t)(r0 + mrow) * DM + kc * 32;
      v16bf a = cvt_ab(xrow + koff, xrow + 16 + koff);
      const __bf16* wrow = woT + (size_t)(nt * 16 + mrow) * DM + kc * 32;
      acc = wmma_bf16(a, load_ab(wrow, koff), acc);
    }
    const int n = nt * 16 + (lane & 15);
#pragma unroll
    for (int r = 0; r < 8; ++r) {
      const int m = r + 8 * (lane >> 4);
      sO[m * 132 + n] = acc[r];
    }
  }
  __syncthreads();

  const int row = threadIdx.x >> 3;
  const int jj  = threadIdx.x & 7;
  float vbuf[16];
  float lsum = 0.f;
#pragma unroll
  for (int i = 0; i < 16; ++i) {
    const int c = jj * 16 + i;
    const float x = sO[row * 132 + c] + bo[c] + Qin[(size_t)(r0 + row) * DM + c];
    vbuf[i] = x;
    lsum += x;
  }
  sStat[row * 8 + jj] = lsum;
  __syncthreads();
  if (jj == 0) {
    float s = 0.f;
#pragma unroll
    for (int t = 0; t < 8; ++t) s += sStat[row * 8 + t];
    sMu[row] = s * (1.0f / 128.0f);
  }
  __syncthreads();
  const float mu = sMu[row];
  float lvar = 0.f;
#pragma unroll
  for (int i = 0; i < 16; ++i) {
    const float d = vbuf[i] - mu;
    lvar += d * d;
  }
  sStat[row * 8 + jj] = lvar;
  __syncthreads();
  if (jj == 0) {
    float s = 0.f;
#pragma unroll
    for (int t = 0; t < 8; ++t) s += sStat[row * 8 + t];
    sRs[row] = rsqrtf(s * (1.0f / 128.0f) + 1e-5f);
  }
  __syncthreads();
  const float rs = sRs[row];
#pragma unroll
  for (int i = 0; i < 16; ++i) {
    const int c = jj * 16 + i;
    out[(size_t)(r0 + row) * DM + c] = (vbuf[i] - mu) * rs * gamma[c] + beta[c];
  }
}

// ---------------------------------------------------------------------------
extern "C" void kernel_launch(void* const* d_in, const int* in_sizes, int n_in,
                              void* d_out, int out_size, void* d_ws,
                              size_t ws_size, hipStream_t stream) {
  const float* Q  = (const float*)d_in[0];
  const float* K  = (const float*)d_in[1];
  const float* V  = (const float*)d_in[2];
  const unsigned char* mask = (const unsigned char*)d_in[3];
  const float* wq = (const float*)d_in[4];
  const float* bq = (const float*)d_in[5];
  const float* wk = (const float*)d_in[6];
  const float* bk = (const float*)d_in[7];
  const float* wv = (const float*)d_in[8];
  const float* bv = (const float*)d_in[9];
  const float* wo = (const float*)d_in[10];
  const float* bo = (const float*)d_in[11];
  const float* gamma = (const float*)d_in[12];
  const float* beta  = (const float*)d_in[13];

  // workspace layout (bytes)
  char* ws = (char*)d_ws;
  const size_t rows = (size_t)B_SZ * S_LEN;          // 16384
  __bf16* wT    = (__bf16*)ws;                        // 4*16384 bf16 = 128 KB
  __bf16* q_ws  = (__bf16*)(ws + 131072);             // rows*32*NH bf16 = 4 MB
  __bf16* k_ws  = (__bf16*)(ws + 131072 + 4194304);
  __bf16* vT_ws = (__bf16*)(ws + 131072 + 2 * 4194304);
  float*  ctx   = (float*) (ws + 131072 + 3 * 4194304); // 8 MB fp32

  float* out  = (float*)d_out;
  float* attn = out + rows * DM;                      // tuple part 2

  hipLaunchKernelGGL(prep_weights, dim3(256), dim3(256), 0, stream,
                     wq, wk, wv, wo, wT);
  hipLaunchKernelGGL(qkv_proj, dim3(rows / 16), dim3(128), 0, stream,
                     Q, K, V, wT, bq, bk, bv, q_ws, k_ws, vT_ws);
  hipLaunchKernelGGL(attn_kernel, dim3(B_SZ * NH * (S_LEN / 16)), dim3(128), 0,
                     stream, q_ws, k_ws, vT_ws, mask, attn, ctx);
  hipLaunchKernelGGL(out_ln, dim3(rows / 16), dim3(128), 0, stream,
                     ctx, wT + 3 * 16384, bo, Q, gamma, beta, out);
}
